// KacLayer_33011118637228
// MI455X (gfx1250) — compile-verified, matrix operand order checked by reading, not run on hardware
//
#include <hip/hip_runtime.h>
#include <stdint.h>

#define DIM     4096
#define NROWS   512
#define NSTEPS  (12 * 4096)
#define KAC_R   16

// GEMM tiling
#define BLK_M   32                 // block M (2 wave rows)
#define BLK_N   256                // block N (4 wave cols x 64)
#define KB      32                 // K per step (one 16x16x32 WMMA chain)
#define LDS_ROW_BYTES 80           // 64B of K data + 16B TDM pad -> aligned, conflict-free
#define TILE_BYTES (BLK_N * LDS_ROW_BYTES)   // 20480 B per (buf, hi/lo)
#define NKSTEP  (DIM / KB)

typedef __attribute__((ext_vector_type(16))) __bf16 v16bf;
typedef __attribute__((ext_vector_type(8)))  float  v8f;

union Frag {
    uint4 u[2];
    v16bf v;
};

#if defined(__gfx1250__) && __has_builtin(__builtin_amdgcn_tensor_load_to_lds)
#define USE_TDM 1
#else
#define USE_TDM 0
#endif

#if USE_TDM
typedef __attribute__((ext_vector_type(4))) unsigned v4u;
typedef __attribute__((ext_vector_type(8))) int      v8i;
typedef __attribute__((ext_vector_type(4))) int      v4i;

// Issue one TDM: copy a [BLK_N x KB] bf16 tile (row stride DIM) from global
// into LDS at lds_byte_off, padding 16B after every 64B row (LDS_ROW_BYTES=80).
__device__ __forceinline__
void tdm_load_tile(const __bf16* gsrc, unsigned lds_byte_off) {
    uint64_t ga = (uint64_t)(uintptr_t)gsrc;
    v4u g0;
    g0[0] = 1u;                                               // count=1, user mode
    g0[1] = lds_byte_off;                                     // lds_addr
    g0[2] = (unsigned)(ga & 0xFFFFFFFFu);                     // global_addr[31:0]
    g0[3] = (unsigned)((ga >> 32) & 0x01FFFFFFu) | (2u << 30);// addr[56:32] | type=2
    v8i g1;
    g1[0] = (1 << 16)        // data_size = 2 bytes
          | (1 << 20)        // pad_enable
          | (3 << 22)        // pad_interval: 16 dwords (=one 64B row)
          | (3 << 25);       // pad_amount: 4 dwords (16B)
    g1[1] = (DIM & 0xFFFF) << 16;                 // tensor_dim0 lo16 at bits[63:48]
    g1[2] = (DIM >> 16) | ((DIM & 0xFFFF) << 16); // tensor_dim0 hi16 | tensor_dim1 lo16
    g1[3] = (DIM >> 16) | (KB << 16);             // tensor_dim1 hi16 | tile_dim0
    g1[4] = BLK_N;                                // tile_dim1 (tile_dim2 = 0)
    g1[5] = DIM;                                  // tensor_dim0_stride lo32
    g1[6] = 0;
    g1[7] = 0;
    v4i z4 = {0, 0, 0, 0};
#if __clang_major__ >= 23
    v8i z8 = {0, 0, 0, 0, 0, 0, 0, 0};
    __builtin_amdgcn_tensor_load_to_lds(g0, g1, z4, z4, z8, 0);
#else
    __builtin_amdgcn_tensor_load_to_lds(g0, g1, z4, z4, 0);
#endif
}
#endif // USE_TDM

// ---------------------------------------------------------------------------
__global__ __launch_bounds__(256)
void split_bf16_kernel(const float* __restrict__ src,
                       __bf16* __restrict__ hi, __bf16* __restrict__ lo, int n) {
    int idx = blockIdx.x * blockDim.x + threadIdx.x;
    if (idx < n) {
        float f = src[idx];
        __bf16 h = (__bf16)f;
        hi[idx] = h;
        lo[idx] = (__bf16)(f - (float)h);
    }
}

// ---------------------------------------------------------------------------
// out[m][n] = sum_k x[m][k] * W[n][k] + b[n], bf16 hi/lo split (3 WMMAs / step).
// Block tile 32M x 256N (8 waves, each 16x64). B tiles staged into LDS by the
// Tensor Data Mover (double-buffered, TENSORcnt-synchronized); A read direct.
// LDS layout: [buf][hi|lo][BLK_N rows][LDS_ROW_BYTES]; all buffer addresses are
// computed arithmetically (no LDS-pointer aggregates -> no static initializers).
// ---------------------------------------------------------------------------
__global__ __launch_bounds__(256)
void wmma_gemm_kernel(const __bf16* __restrict__ Ahi, const __bf16* __restrict__ Alo,
                      const __bf16* __restrict__ Bhi, const __bf16* __restrict__ Blo,
                      const float* __restrict__ bias, float* __restrict__ out) {
    extern __shared__ char ldsraw[];   // 4 * TILE_BYTES = 80 KB
    const int tid  = threadIdx.x;
    const int lane = tid & 31;
    const int wave = tid >> 5;
    const int lrow = lane & 15;
    const int half = lane >> 4;
    const int koff = half << 3;                         // ISA 16-bit A/B K sub-offset
    const int mbase  = blockIdx.y * BLK_M + (wave >> 2) * 16;
    const int nblock = blockIdx.x * BLK_N;
    const int nwave  = (wave & 3) * 64;

#if USE_TDM
    const unsigned ldsBase = (unsigned)(uintptr_t)(void*)ldsraw; // LDS aperture keeps offset in addr[31:0]
#endif

    v8f acc[4];
#pragma unroll
    for (int c = 0; c < 4; ++c)
#pragma unroll
        for (int r = 0; r < 8; ++r) acc[c][r] = 0.0f;

    const __bf16* aHiRow = Ahi + (size_t)(mbase + lrow) * DIM;
    const __bf16* aLoRow = Alo + (size_t)(mbase + lrow) * DIM;

    auto stage = [&](int buf, int k0) {
        const unsigned hiOff = (unsigned)buf * 2u * TILE_BYTES;
#if USE_TDM
        if (tid < 32) {   // one wave issues the DMA (TDM ignores EXEC; branch is wave-uniform)
            tdm_load_tile(Bhi + (size_t)nblock * DIM + k0, ldsBase + hiOff);
            tdm_load_tile(Blo + (size_t)nblock * DIM + k0, ldsBase + hiOff + TILE_BYTES);
        }
#else
        // cooperative fallback: thread t copies row t (BLK_N == blockDim.x)
        const uint4* sh = (const uint4*)(Bhi + (size_t)(nblock + tid) * DIM + k0);
        const uint4* sl = (const uint4*)(Blo + (size_t)(nblock + tid) * DIM + k0);
        uint4* dh = (uint4*)(ldsraw + hiOff + tid * LDS_ROW_BYTES);
        uint4* dl = (uint4*)(ldsraw + hiOff + TILE_BYTES + tid * LDS_ROW_BYTES);
#pragma unroll
        for (int i = 0; i < 4; ++i) { dh[i] = sh[i]; dl[i] = sl[i]; }
#endif
    };

    stage(0, 0);
    int cur = 0;
    for (int ks = 0; ks < NKSTEP; ++ks) {
        __syncthreads();                       // everyone done reading buf[cur^1]
        if (ks + 1 < NKSTEP) {
            stage(cur ^ 1, (ks + 1) * KB);
#if USE_TDM
            if (tid < 32) __builtin_amdgcn_s_wait_tensorcnt(2); // buf[cur]'s 2 TDMs done
#endif
        } else {
#if USE_TDM
            if (tid < 32) __builtin_amdgcn_s_wait_tensorcnt(0);
#endif
        }
        __syncthreads();                       // buf[cur] visible to all waves

        const int k0 = ks * KB;
        Frag ah, al;
        ah.u[0] = *(const uint4*)(aHiRow + k0 + koff);
        ah.u[1] = *(const uint4*)(aHiRow + k0 + koff + 16);
        al.u[0] = *(const uint4*)(aLoRow + k0 + koff);
        al.u[1] = *(const uint4*)(aLoRow + k0 + koff + 16);

        const char* bh_base = ldsraw + cur * 2 * TILE_BYTES;
        const char* bl_base = bh_base + TILE_BYTES;
#pragma unroll
        for (int c = 0; c < 4; ++c) {
            const int nloc = nwave + c * 16 + lrow;
            const char* rh = bh_base + nloc * LDS_ROW_BYTES + koff * 2;
            const char* rl = bl_base + nloc * LDS_ROW_BYTES + koff * 2;
            Frag bh, bl;
            bh.u[0] = *(const uint4*)(rh);
            bh.u[1] = *(const uint4*)(rh + 32);
            bl.u[0] = *(const uint4*)(rl);
            bl.u[1] = *(const uint4*)(rl + 32);

            acc[c] = __builtin_amdgcn_wmma_f32_16x16x32_bf16(
                false, ah.v, false, bh.v, (short)0, acc[c], false, false);
            acc[c] = __builtin_amdgcn_wmma_f32_16x16x32_bf16(
                false, ah.v, false, bl.v, (short)0, acc[c], false, false);
            acc[c] = __builtin_amdgcn_wmma_f32_16x16x32_bf16(
                false, al.v, false, bh.v, (short)0, acc[c], false, false);
        }
        cur ^= 1;
    }

    const int mrow0 = mbase + half * 8;
#pragma unroll
    for (int c = 0; c < 4; ++c) {
        int n = nblock + nwave + c * 16 + lrow;
        float bv = bias[n];
#pragma unroll
        for (int r = 0; r < 8; ++r)
            out[(size_t)(mrow0 + r) * DIM + n] = acc[c][r] + bv;
    }
}

// ---------------------------------------------------------------------------
// Kac walk: lanes 0..15 own column i of rows 0..15, lanes 16..31 own column j.
// One ds_load + one ds_bpermute + fma + one ds_store per step; schedule is
// wave-uniform -> SGPRs; DS ops are in-order per wave so the RAW chain holds.
// ---------------------------------------------------------------------------
__device__ __forceinline__
void kac_walk_lds(float* lds, int r, int isJ,
                  const int* __restrict__ gi, const int* __restrict__ gj,
                  const float* __restrict__ gc, const float* __restrict__ gs) {
#pragma unroll 4
    for (int t = 0; t < NSTEPS; ++t) {
        const int   col = isJ ? gj[t] : gi[t];
        const float c   = gc[t];
        const float s   = gs[t];
        float own = lds[col * KAC_R + r];
        float oth = __shfl_xor(own, KAC_R, 32);
        lds[col * KAC_R + r] = c * own + (isJ ? -s : s) * oth;
    }
}

__global__ __launch_bounds__(256)
void kac_kernel(const float* __restrict__ x,
                const int* __restrict__ i1, const int* __restrict__ j1,
                const float* __restrict__ c1, const float* __restrict__ s1,
                const int* __restrict__ i2, const int* __restrict__ j2,
                const float* __restrict__ c2, const float* __restrict__ s2,
                const float* __restrict__ vec, float* __restrict__ out) {
    extern __shared__ float lds[];          // [DIM][KAC_R] column-interleaved, 256 KB
    const int tid = threadIdx.x;
    const int rowbase = blockIdx.x * KAC_R;

    for (int e = tid; e < KAC_R * DIM; e += blockDim.x) {
        int r = e / DIM, c = e % DIM;
        lds[c * KAC_R + r] = x[(size_t)(rowbase + r) * DIM + c];
    }
    __syncthreads();

    if (tid < 32)
        kac_walk_lds(lds, tid & (KAC_R - 1), tid >> 4, i1, j1, c1, s1);
    __syncthreads();

    for (int e = tid; e < KAC_R * DIM; e += blockDim.x)
        lds[e] *= vec[e >> 4];
    __syncthreads();

    if (tid < 32)
        kac_walk_lds(lds, tid & (KAC_R - 1), tid >> 4, i2, j2, c2, s2);
    __syncthreads();

    for (int e = tid; e < KAC_R * DIM; e += blockDim.x) {
        int r = e / DIM, c = e % DIM;
        out[(size_t)(rowbase + r) * DIM + c] += lds[c * KAC_R + r];
    }
}

// ---------------------------------------------------------------------------
extern "C" void kernel_launch(void* const* d_in, const int* in_sizes, int n_in,
                              void* d_out, int out_size, void* d_ws, size_t ws_size,
                              hipStream_t stream) {
    const float* x   = (const float*)d_in[0];
    const float* W   = (const float*)d_in[1];
    const float* b   = (const float*)d_in[2];
    const float* vec = (const float*)d_in[3];
    const int*   i1  = (const int*)d_in[4];
    const int*   j1  = (const int*)d_in[5];
    const float* c1  = (const float*)d_in[6];
    const float* s1  = (const float*)d_in[7];
    const int*   i2  = (const int*)d_in[8];
    const int*   j2  = (const int*)d_in[9];
    const float* c2  = (const float*)d_in[10];
    const float* s2  = (const float*)d_in[11];
    float* out = (float*)d_out;

    // workspace: Whi | Wlo | xhi | xlo  (72 MB total)
    __bf16* Whi = (__bf16*)d_ws;
    __bf16* Wlo = Whi + (size_t)DIM * DIM;
    __bf16* xhi = Wlo + (size_t)DIM * DIM;
    __bf16* xlo = xhi + (size_t)NROWS * DIM;

    const int nW = DIM * DIM;
    const int nX = NROWS * DIM;
    split_bf16_kernel<<<(nW + 255) / 256, 256, 0, stream>>>(W, Whi, Wlo, nW);
    split_bf16_kernel<<<(nX + 255) / 256, 256, 0, stream>>>(x, xhi, xlo, nX);

    wmma_gemm_kernel<<<dim3(DIM / BLK_N, NROWS / BLK_M), 256,
                       4 * TILE_BYTES, stream>>>(xhi, xlo, Whi, Wlo, b, out);

    kac_kernel<<<NROWS / KAC_R, 256, KAC_R * DIM * sizeof(float), stream>>>(
        x, i1, j1, c1, s1, i2, j2, c2, s2, vec, out);
}
